// GraphConvolution_56186762166661
// MI455X (gfx1250) — matrix operand order repaired and checked by reading, not compiled, not run on hardware
//
#include <hip/hip_runtime.h>
#include <hip/hip_bf16.h>

#define DIM 128
#define PITCH 129    // f32 LDS pitch (conflict-free row walks)
#define BPITCH 136   // bf16 LDS pitch (keeps 16B alignment per row)
#define TILE_M 128

typedef __bf16 bf16_t;
typedef bf16_t v16bf __attribute__((ext_vector_type(16)));
typedef unsigned short v16us __attribute__((ext_vector_type(16)));
typedef float v8f __attribute__((ext_vector_type(8)));

union BfFrag {
    v16us u;
    v16bf b;
    float4 q[2];  // two 16B halves for raw LDS b128 loads
};

__device__ __forceinline__ unsigned short f32_to_bf16_rne(float f) {
    unsigned u = __float_as_uint(f);
    unsigned r = u + 0x7fffu + ((u >> 16) & 1u);
    return (unsigned short)(r >> 16);
}
__device__ __forceinline__ float bf16_to_f32(unsigned short h) {
    return __uint_as_float(((unsigned)h) << 16);
}

// A fragment: 16x32 (MxK) bf16 from f32 LDS (pitch PITCH).  lane = 16*g + m.
// VGPR v, half j:  k = 16*(v>=4) + 2*(v&3) + 8*g + j  -> elements ascend in k.
__device__ __forceinline__ void load_a_frag(const float* __restrict__ s,
                                            int m_base, int k_base, int lane,
                                            BfFrag& hi, BfFrag& lo) {
    const int g = lane >> 4, m = lane & 15;
    const float* row = s + (m_base + m) * PITCH + k_base + 8 * g;
#pragma unroll
    for (int v = 0; v < 8; ++v) {
        const int k0 = ((v >= 4) ? 16 : 0) + 2 * (v & 3);
#pragma unroll
        for (int j = 0; j < 2; ++j) {
            float a = row[k0 + j];
            unsigned short h = f32_to_bf16_rne(a);
            hi.u[2 * v + j] = h;
            lo.u[2 * v + j] = f32_to_bf16_rne(a - bf16_to_f32(h));
        }
    }
}

// B fragment: 32x16 (KxN) bf16, element s: k = 8*g + s (s<8), 16 + 8*g + (s-8) (s>=8).
// Source is pre-converted TRANSPOSED bf16 LDS: sWT[n * BPITCH + k].
// => one lane reads two runs of 8 bf16, 16 apart = two 16B loads (ds_load_b128).
__device__ __forceinline__ void load_b_frag(const unsigned short* __restrict__ sWT,
                                            int k_base, int n_base, int lane,
                                            BfFrag& f) {
    const int g = lane >> 4, nn = lane & 15;
    const unsigned short* p = sWT + (n_base + nn) * BPITCH + k_base + 8 * g;
    f.q[0] = *(const float4*)(p);
    f.q[1] = *(const float4*)(p + 16);
}

// Cooperative: convert global f32 [k][n] weight into transposed bf16 hi/lo LDS.
__device__ __forceinline__ void stage_w_T(const float* __restrict__ gW,
                                          unsigned short* __restrict__ hi,
                                          unsigned short* __restrict__ lo, int tid) {
    for (int i = tid; i < DIM * DIM; i += 256) {
        int k = i >> 7, nn = i & (DIM - 1);
        float a = gW[i];
        unsigned short h = f32_to_bf16_rne(a);
        hi[nn * BPITCH + k] = h;
        lo[nn * BPITCH + k] = f32_to_bf16_rne(a - bf16_to_f32(h));
    }
}

__device__ __forceinline__ v8f wmma_bf16(const BfFrag& a, const BfFrag& b, v8f c) {
    return __builtin_amdgcn_wmma_f32_16x16x32_bf16(
        /*neg_a=*/false, a.b, /*neg_b=*/false, b.b,
        /*c_mod=*/(short)0, c, /*reuse_a=*/false, /*reuse_b=*/false);
}

// 16x128 row-slab GEMM for one wave: D(slab) += A(slab,:128) * B(128x128),
// bf16x3 split (hi*hi + hi*lo + lo*hi), fp32 accumulate, result -> f32 LDS.
__device__ __forceinline__ void gemm_slab(const float* __restrict__ srcA,
                                          const unsigned short* __restrict__ sWhi,
                                          const unsigned short* __restrict__ sWlo,
                                          float* __restrict__ dst,
                                          int m_base, int lane) {
    BfFrag a_hi[4], a_lo[4];
#pragma unroll
    for (int ks = 0; ks < 4; ++ks)
        load_a_frag(srcA, m_base, ks * 32, lane, a_hi[ks], a_lo[ks]);

#pragma unroll
    for (int nt = 0; nt < 8; ++nt) {
        v8f acc = {0.f, 0.f, 0.f, 0.f, 0.f, 0.f, 0.f, 0.f};
#pragma unroll
        for (int ks = 0; ks < 4; ++ks) {
            BfFrag b_hi, b_lo;
            load_b_frag(sWhi, ks * 32, nt * 16, lane, b_hi);
            load_b_frag(sWlo, ks * 32, nt * 16, lane, b_lo);
            acc = wmma_bf16(a_hi[ks], b_hi, acc);
            acc = wmma_bf16(a_hi[ks], b_lo, acc);
            acc = wmma_bf16(a_lo[ks], b_hi, acc);
        }
        const int g = lane >> 4, nn = nt * 16 + (lane & 15);
#pragma unroll
        for (int r = 0; r < 8; ++r)
            dst[(m_base + r + 8 * g) * PITCH + nn] = acc[r];
    }
}

// ---------------------------------------------------------------------------
// Kernel 1: pre = sigmoid(1 + 0.1*norm((x@W0)@glu_W)) * (x@W0)
// ---------------------------------------------------------------------------
__global__ __launch_bounds__(256) void k_gemm_glu(
    const float* __restrict__ x, const float* __restrict__ W0,
    const float* __restrict__ glu_W, const float* __restrict__ glu_scale,
    float* __restrict__ pre_out, int n) {
    __shared__ __align__(16) float sXG[TILE_M * PITCH];          // x tile, then gate slab
    __shared__ __align__(16) unsigned short sWhi[DIM * BPITCH];  // W hi (transposed)
    __shared__ __align__(16) unsigned short sWlo[DIM * BPITCH];  // W lo (transposed)
    __shared__ __align__(16) float sP[TILE_M * PITCH];           // pre_sup tile (f32)
    __shared__ float sMu[TILE_M];
    __shared__ float sRs[TILE_M];

    const int tid = threadIdx.x;
    const int wave = tid >> 5;
    const int lane = tid & 31;
    const int row0 = blockIdx.x * TILE_M;
    const int m_base = wave * 16;

    for (int i = tid; i < TILE_M * DIM; i += 256) {
        int r = i >> 7, c = i & (DIM - 1);
        int gr = row0 + r;
        if (gr >= n) gr = n - 1;  // clamp; junk rows discarded at store
        sXG[r * PITCH + c] = x[(size_t)gr * DIM + c];
    }
    stage_w_T(W0, sWhi, sWlo, tid);
    __syncthreads();

    // phase 1: pre_sup = x @ W0
    gemm_slab(sXG, sWhi, sWlo, sP, m_base, lane);
    __syncthreads();

    // phase 2: gate = pre_sup @ glu_W  (restage weights; gate overwrites sXG)
    stage_w_T(glu_W, sWhi, sWlo, tid);
    __syncthreads();
    gemm_slab(sP, sWhi, sWlo, sXG, m_base, lane);
    __syncthreads();

    // per-row layernorm stats of gate
    if (tid < TILE_M) {
        const float* rowp = &sXG[tid * PITCH];
        float mu = 0.f;
        for (int c = 0; c < DIM; ++c) mu += rowp[c];
        mu *= (1.0f / DIM);
        float var = 0.f;
        for (int c = 0; c < DIM; ++c) {
            float d = rowp[c] - mu;
            var += d * d;
        }
        var *= (1.0f / DIM);
        sMu[tid] = mu;
        sRs[tid] = rsqrtf(var + 1e-20f);
    }
    __syncthreads();

    const float sc = 0.1f * fmaxf(glu_scale[0], 1e-4f);
    for (int i = tid; i < TILE_M * DIM; i += 256) {
        int r = i >> 7, c = i & (DIM - 1);
        int gr = row0 + r;
        if (gr < n) {
            float gate = sXG[r * PITCH + c];
            float gn = 1.0f + (gate - sMu[r]) * sRs[r] * sc;
            float sig = 1.0f / (1.0f + __expf(-gn));
            pre_out[(size_t)gr * DIM + c] = sig * sP[r * PITCH + c];
        }
    }
}

// ---------------------------------------------------------------------------
// Kernel 2: zero the scatter accumulator
// ---------------------------------------------------------------------------
__global__ __launch_bounds__(256) void k_zero(float4* __restrict__ p, long long n4) {
    long long i = (long long)blockIdx.x * blockDim.x + threadIdx.x;
    long long stride = (long long)gridDim.x * blockDim.x;
    float4 z = make_float4(0.f, 0.f, 0.f, 0.f);
    for (; i < n4; i += stride) p[i] = z;
}

// ---------------------------------------------------------------------------
// Kernel 3: SpMM scatter.  One wave per edge; 32 lanes x float4 = 128 feats.
// pre (51MB) + agg (51MB) are L2-resident (192MB), so this runs at L2 speed.
// ---------------------------------------------------------------------------
__global__ __launch_bounds__(256) void k_spmm(
    const int* __restrict__ rows, const int* __restrict__ cols,
    const float* __restrict__ vals, const float* __restrict__ pre,
    float* __restrict__ agg, int nE) {
    const int warp = (int)((blockIdx.x * blockDim.x + threadIdx.x) >> 5);
    const int lane = threadIdx.x & 31;
    const int nW = (int)((gridDim.x * blockDim.x) >> 5);
    for (int e = warp; e < nE; e += nW) {
        if (e + nW < nE) {  // global_prefetch_b8 on the index streams
            __builtin_prefetch(cols + e + nW, 0, 0);
            __builtin_prefetch(rows + e + nW, 0, 0);
        }
        const int c = cols[e];
        const int r = rows[e];
        const float v = vals[e];
        float4 m = *((const float4*)(pre + (size_t)c * DIM) + lane);
        float* dst = agg + (size_t)r * DIM + (size_t)lane * 4;
        atomicAdd(dst + 0, v * m.x);  // no-return global_atomic_add_f32
        atomicAdd(dst + 1, v * m.y);
        atomicAdd(dst + 2, v * m.z);
        atomicAdd(dst + 3, v * m.w);
    }
}

// ---------------------------------------------------------------------------
// Kernel 4: out = norm(relu(agg) @ W_skip)*s1 + norm(x)*s2
// ---------------------------------------------------------------------------
__global__ __launch_bounds__(256) void k_out(
    const float* __restrict__ agg, const float* __restrict__ x,
    const float* __restrict__ W_skip, const float* __restrict__ ln_scale,
    const float* __restrict__ ln_scale2, float* __restrict__ out, int n) {
    __shared__ __align__(16) float sA[TILE_M * PITCH];           // relu(agg) tile
    __shared__ __align__(16) unsigned short sWhi[DIM * BPITCH];  // W_skip hi (T)
    __shared__ __align__(16) unsigned short sWlo[DIM * BPITCH];  // W_skip lo (T)
    __shared__ __align__(16) float sY[TILE_M * PITCH];           // GEMM result
    __shared__ float sMuY[TILE_M], sRsY[TILE_M];
    __shared__ float sMuX[TILE_M], sRsX[TILE_M];

    const int tid = threadIdx.x;
    const int wave = tid >> 5;
    const int lane = tid & 31;
    const int row0 = blockIdx.x * TILE_M;
    const int m_base = wave * 16;

    for (int i = tid; i < TILE_M * DIM; i += 256) {
        int r = i >> 7, c = i & (DIM - 1);
        int gr = row0 + r;
        if (gr >= n) gr = n - 1;
        sA[r * PITCH + c] = fmaxf(agg[(size_t)gr * DIM + c], 0.f);
    }
    stage_w_T(W_skip, sWhi, sWlo, tid);
    __syncthreads();

    gemm_slab(sA, sWhi, sWlo, sY, m_base, lane);
    __syncthreads();

    if (tid < TILE_M) {
        const float* rowp = &sY[tid * PITCH];
        float mu = 0.f;
        for (int c = 0; c < DIM; ++c) mu += rowp[c];
        mu *= (1.0f / DIM);
        float var = 0.f;
        for (int c = 0; c < DIM; ++c) {
            float d = rowp[c] - mu;
            var += d * d;
        }
        var *= (1.0f / DIM);
        sMuY[tid] = mu;
        sRsY[tid] = rsqrtf(var + 1e-20f);

        int gr = row0 + tid;
        if (gr >= n) gr = n - 1;
        const float* xr = x + (size_t)gr * DIM;
        float mux = 0.f;
        for (int c = 0; c < DIM; ++c) mux += xr[c];
        mux *= (1.0f / DIM);
        float varx = 0.f;
        for (int c = 0; c < DIM; ++c) {
            float d = xr[c] - mux;
            varx += d * d;
        }
        varx *= (1.0f / DIM);
        sMuX[tid] = mux;
        sRsX[tid] = rsqrtf(varx + 1e-20f);
    }
    __syncthreads();

    const float s1 = fmaxf(ln_scale[0], 1e-4f);
    const float s2 = fmaxf(ln_scale2[0], 1e-4f);
    for (int i = tid; i < TILE_M * DIM; i += 256) {
        int r = i >> 7, c = i & (DIM - 1);
        int gr = row0 + r;
        if (gr < n) {
            float y = (sY[r * PITCH + c] - sMuY[r]) * sRsY[r] * s1;
            float xv = x[(size_t)gr * DIM + c];
            float xn = (xv - sMuX[r]) * sRsX[r] * s2;
            out[(size_t)gr * DIM + c] = y + xn;
        }
    }
}

// ---------------------------------------------------------------------------
extern "C" void kernel_launch(void* const* d_in, const int* in_sizes, int n_in,
                              void* d_out, int out_size, void* d_ws, size_t ws_size,
                              hipStream_t stream) {
    const float* x = (const float*)d_in[0];
    const int* erow = (const int*)d_in[1];
    const int* ecol = (const int*)d_in[2];
    const float* evals = (const float*)d_in[3];
    const float* W0 = (const float*)d_in[4];
    const float* gluW = (const float*)d_in[5];
    const float* glu_s = (const float*)d_in[6];
    const float* ln_s = (const float*)d_in[7];
    const float* ln_s2 = (const float*)d_in[8];
    const float* Wskip = (const float*)d_in[9];
    float* out = (float*)d_out;

    const int n = in_sizes[0] / DIM;
    const int nE = in_sizes[1];

    float* pre = (float*)d_ws;           // n*128 f32 (gated pre_sup)
    float* agg = pre + (size_t)n * DIM;  // n*128 f32 (scatter accumulator)

    const int nTiles = (n + TILE_M - 1) / TILE_M;
    const long long n4 = (long long)n * DIM / 4;

    k_zero<<<2048, 256, 0, stream>>>((float4*)agg, n4);
    k_gemm_glu<<<nTiles, 256, 0, stream>>>(x, W0, gluW, glu_s, pre, n);
    k_spmm<<<2048, 256, 0, stream>>>(erow, ecol, evals, pre, agg, nE);
    k_out<<<nTiles, 256, 0, stream>>>(agg, x, Wskip, ln_s, ln_s2, out, n);
}
